// GraphProcessor_8065948582592
// MI455X (gfx1250) — compile-verified, matrix operand order checked by reading, not run on hardware
//
#include <hip/hip_runtime.h>
#include <hip/hip_bf16.h>

// ---------------- problem constants ----------------
constexpr int NN   = 32768;   // N_NODES
constexpr int NA   = 28672;   // N_ACTIVE
constexpr int NE   = 229376;  // N_EDGES
constexpr int C    = 64;
constexpr int ECH  = 2;
constexpr int HID  = 128;
constexpr int NL   = 8;

constexpr int K_EW1 = 160;    // 130 padded to 160
constexpr int K_N1  = 96;     // 66 padded to 96
constexpr int NPAD_N1W2 = 80; // 66 padded to 80 rows of Bt

typedef __attribute__((ext_vector_type(16))) _Float16 v16h;
typedef __attribute__((ext_vector_type(8)))  float    v8f;
typedef __attribute__((ext_vector_type(8)))  unsigned int v8u;

// ---------------- fragment load (16-bit A/B layout) ----------------
// lane<16: K k..k+7 (VGPR0-3) and k+16..k+23 (VGPR4-7); lane>=16: +8.
// caller passes p = rowBase + k + ((lane>=16)?8:0); loads p[0..7], p[16..23].
__device__ __forceinline__ v16h ldfrag16(const _Float16* p) {
    const uint4* q = (const uint4*)p;   // 16B-aligned by construction
    uint4 lo = q[0];
    uint4 hi = q[2];                    // +32 bytes = +16 halves
    v8u r;
    r[0] = lo.x; r[1] = lo.y; r[2] = lo.z; r[3] = lo.w;
    r[4] = hi.x; r[5] = hi.y; r[6] = hi.z; r[7] = hi.w;
    return __builtin_bit_cast(v16h, r);
}

// ---------------- generic WMMA GEMM: C[M,N] = act(A[M,K] * Bt[N,K]^T + bias) --------------
// one wave per 16x(16*NTW) strip of C: A fragment is loaded once per k-step and
// reused against NTW weight fragments (weights are WGP$-resident), cutting
// activation re-reads from L2 by NTW x. 8 waves/block; M multiple of 16 guaranteed.
template<int NTW, bool RELU, bool OUT_F16, bool ACCUM>
__global__ void __launch_bounds__(256)
wmma_gemm(const _Float16* __restrict__ A, int lda,
          const _Float16* __restrict__ Bt, int ldb,
          void* __restrict__ Cv, int ldc, int Nact,
          const float* __restrict__ bias, int K)
{
    const int wave  = threadIdx.x >> 5;
    const int lane  = threadIdx.x & 31;
    const int mTile = blockIdx.y * 8 + wave;
    const int nBase = blockIdx.x * NTW;
    const int h16   = lane >> 4;
    const int l15   = lane & 15;
    const int koff  = h16 ? 8 : 0;
    const long arow = (long)(mTile * 16 + l15) * lda;
    const _Float16* bbase = Bt + (long)(nBase * 16 + l15) * ldb;

    v8f c[NTW] = {};
    if (ACCUM) {
        const float* Cp = (const float*)Cv;
#pragma unroll
        for (int t = 0; t < NTW; ++t) {
            const int ncol = (nBase + t) * 16 + l15;
            if (ncol < Nact) {
#pragma unroll
                for (int r = 0; r < 8; ++r)
                    c[t][r] = Cp[(long)(mTile * 16 + h16 * 8 + r) * ldc + ncol];
            }
        }
    }

    for (int k = 0; k < K; k += 32) {
        const v16h a = ldfrag16(A + arow + k + koff);
#pragma unroll
        for (int t = 0; t < NTW; ++t) {
            const v16h b = ldfrag16(bbase + (long)t * 16 * ldb + k + koff);
            c[t] = __builtin_amdgcn_wmma_f32_16x16x32_f16(false, a, false, b,
                                                          (short)0, c[t], false, false);
        }
    }

#pragma unroll
    for (int t = 0; t < NTW; ++t) {
        const int ncol = (nBase + t) * 16 + l15;
        if (ncol < Nact) {
            const float bv = bias ? bias[ncol] : 0.f;
#pragma unroll
            for (int r = 0; r < 8; ++r) {
                float v = c[t][r] + bv;
                if (RELU) v = fmaxf(v, 0.f);
                const long m = (long)(mTile * 16 + h16 * 8 + r);
                if (OUT_F16) ((_Float16*)Cv)[m * ldc + ncol] = (_Float16)v;
                else         ((float*)Cv)[m * ldc + ncol]    = v;
            }
        }
    }
}

// ---------------- weight prep: W[L,K,N] f32 -> Wt[L,Npad,Kpad] f16 (transposed, zero-padded) ----------------
__global__ void prep_weight(const float* __restrict__ W, _Float16* __restrict__ Wt,
                            int Kact, int Nact, int Kpad, int Npad)
{
    const int t = blockIdx.x * blockDim.x + threadIdx.x;
    const int l = blockIdx.y;
    if (t >= Npad * Kpad) return;
    const int n = t / Kpad, k = t % Kpad;
    float v = (k < Kact && n < Nact) ? W[((long)l * Kact + k) * Nact + n] : 0.f;
    Wt[((long)l * Npad + n) * Kpad + k] = (_Float16)v;
}

// ---------------- misc utility kernels ----------------
__global__ void init_nodes(const float* __restrict__ x, float* __restrict__ nodes) {
    const long i = (long)blockIdx.x * blockDim.x + threadIdx.x;
    if (i >= (long)NN * C) return;
    const long pad = (long)(NN - NA) * C;
    nodes[i] = (i < pad) ? 0.f : x[i - pad];
}
__global__ void copy_f32(const float* __restrict__ s, float* __restrict__ d, long n) {
    const long i = (long)blockIdx.x * blockDim.x + threadIdx.x;
    if (i < n) d[i] = s[i];
}
__global__ void zero_f32(float* __restrict__ d, long n) {
    const long i = (long)blockIdx.x * blockDim.x + threadIdx.x;
    if (i < n) d[i] = 0.f;
}
__global__ void count_deg(const int* __restrict__ col, float* __restrict__ cnt) {
    const int e = blockIdx.x * blockDim.x + threadIdx.x;
    if (e < NE) atomicAdd(&cnt[col[e]], 1.f);
}
__global__ void clamp_cnt(float* __restrict__ cnt) {
    const int i = blockIdx.x * blockDim.x + threadIdx.x;
    if (i < NN) cnt[i] = fmaxf(cnt[i], 1.f);
}

// build Xe[E,160] = f16(concat(nodes[row], nodes[col], edges)), zero-pad 130..159
__global__ void build_edge_input(const float* __restrict__ nodes, const float* __restrict__ edges,
                                 const int* __restrict__ row, const int* __restrict__ col,
                                 _Float16* __restrict__ Xe)
{
    const long t = (long)blockIdx.x * blockDim.x + threadIdx.x;
    const long e = t >> 5; const int lane = (int)(t & 31);
    if (e >= NE) return;
    const int r = row[e], cc = col[e];
    _Float16* out = Xe + e * K_EW1;
    for (int j = lane; j < C; j += 32) {
        out[j]     = (_Float16)nodes[(long)r  * C + j];
        out[C + j] = (_Float16)nodes[(long)cc * C + j];
    }
    if (lane < ECH) out[2 * C + lane] = (_Float16)edges[e * ECH + lane];
    for (int j = 2 * C + ECH + lane; j < K_EW1; j += 32) out[j] = (_Float16)0.f;
}

// edge head: y = H*w2 + b2 + Xe*resw + resb ; LN(2) ; + edges ; emit edges & Xn[E,96]
__global__ void edge_out(const _Float16* __restrict__ Xe, const _Float16* __restrict__ H,
                         const float* __restrict__ w2,  const float* __restrict__ b2,
                         const float* __restrict__ rw,  const float* __restrict__ rb,
                         const float* __restrict__ g,   const float* __restrict__ bln,
                         float* __restrict__ edges, _Float16* __restrict__ Xn)
{
    const int e = blockIdx.x * blockDim.x + threadIdx.x;
    if (e >= NE) return;
    const _Float16* h = H  + (long)e * HID;
    const _Float16* x = Xe + (long)e * K_EW1;
    float y0 = b2[0], y1 = b2[1];
    for (int j = 0; j < HID; ++j) { float hv = (float)h[j]; y0 += hv * w2[2*j]; y1 += hv * w2[2*j+1]; }
    float r0 = rb[0], r1 = rb[1];
    for (int j = 0; j < 2 * C + ECH; ++j) { float xv = (float)x[j]; r0 += xv * rw[2*j]; r1 += xv * rw[2*j+1]; }
    const float s0 = y0 + r0, s1 = y1 + r1;
    const float mean = 0.5f * (s0 + s1);
    const float d0 = s0 - mean, d1 = s1 - mean;
    const float inv = rsqrtf(0.5f * (d0 * d0 + d1 * d1) + 1e-5f);
    const float o0 = d0 * inv * g[0] + bln[0] + edges[e * ECH];
    const float o1 = d1 * inv * g[1] + bln[1] + edges[e * ECH + 1];
    edges[e * ECH] = o0; edges[e * ECH + 1] = o1;
    _Float16* xn = Xn + (long)e * K_N1;
    for (int j = 0; j < C; ++j) xn[j] = x[j];
    xn[C] = (_Float16)o0; xn[C + 1] = (_Float16)o1;
    for (int j = C + ECH; j < K_N1; ++j) xn[j] = (_Float16)0.f;
}

// m = LN(Xn66 + M) ; atomic scatter-add onto agg[col]
__global__ void scatter_mean_ln(const float* __restrict__ M, const _Float16* __restrict__ Xn,
                                const int* __restrict__ col,
                                const float* __restrict__ g, const float* __restrict__ bln,
                                float* __restrict__ agg)
{
    const int e = blockIdx.x * blockDim.x + threadIdx.x;
    if (e >= NE) return;
    const float* m = M + (long)e * (C + ECH);
    const _Float16* x = Xn + (long)e * K_N1;
    float v[C + ECH]; float mean = 0.f;
#pragma unroll
    for (int j = 0; j < C + ECH; ++j) { v[j] = m[j] + (float)x[j]; mean += v[j]; }
    mean *= (1.f / (C + ECH));
    float var = 0.f;
#pragma unroll
    for (int j = 0; j < C + ECH; ++j) { float d = v[j] - mean; var += d * d; }
    const float inv = rsqrtf(var * (1.f / (C + ECH)) + 1e-5f);
    const int cc = col[e];
#pragma unroll
    for (int j = 0; j < C + ECH; ++j)
        atomicAdd(&agg[(long)cc * (C + ECH) + j], (v[j] - mean) * inv * g[j] + bln[j]);
}

// Xn2[N,160] = f16(concat(nodes, agg/cnt)), zero-pad tail
__global__ void build_node2_input(const float* __restrict__ nodes, const float* __restrict__ agg,
                                  const float* __restrict__ cnt, _Float16* __restrict__ Xn2)
{
    const long t = (long)blockIdx.x * blockDim.x + threadIdx.x;
    const long n = t >> 5; const int lane = (int)(t & 31);
    if (n >= NN) return;
    const float ic = 1.f / cnt[n];
    _Float16* out = Xn2 + n * K_EW1;
    for (int j = lane; j < C; j += 32) out[j] = (_Float16)nodes[n * C + j];
    for (int j = lane; j < C + ECH; j += 32) out[C + j] = (_Float16)(agg[n * (C + ECH) + j] * ic);
    for (int j = 2 * C + ECH + lane; j < K_EW1; j += 32) out[j] = (_Float16)0.f;
}

// nodes += LN(T)  (T already holds h+res with both biases)
__global__ void node_out(const float* __restrict__ T, float* __restrict__ nodes,
                         const float* __restrict__ g, const float* __restrict__ bln)
{
    const int n = blockIdx.x * blockDim.x + threadIdx.x;
    if (n >= NN) return;
    const float* t = T + (long)n * C;
    float v[C]; float mean = 0.f;
#pragma unroll
    for (int j = 0; j < C; ++j) { v[j] = t[j]; mean += v[j]; }
    mean *= (1.f / C);
    float var = 0.f;
#pragma unroll
    for (int j = 0; j < C; ++j) { float d = v[j] - mean; var += d * d; }
    const float inv = rsqrtf(var * (1.f / C) + 1e-5f);
#pragma unroll
    for (int j = 0; j < C; ++j)
        nodes[(long)n * C + j] += (v[j] - mean) * inv * g[j] + bln[j];
}

// ---------------- host launcher ----------------
extern "C" void kernel_launch(void* const* d_in, const int* in_sizes, int n_in,
                              void* d_out, int out_size, void* d_ws, size_t ws_size,
                              hipStream_t stream)
{
    (void)in_sizes; (void)n_in; (void)out_size; (void)ws_size;
    const float* x        = (const float*)d_in[0];
    const int*   eidx     = (const int*)  d_in[1];
    const float* eattr    = (const float*)d_in[2];
    const float* e_res_w  = (const float*)d_in[3];
    const float* e_res_b  = (const float*)d_in[4];
    const float* e_w1     = (const float*)d_in[5];
    const float* e_b1     = (const float*)d_in[6];
    const float* e_w2     = (const float*)d_in[7];
    const float* e_b2     = (const float*)d_in[8];
    const float* e_ln_g   = (const float*)d_in[9];
    const float* e_ln_b   = (const float*)d_in[10];
    const float* n1_w1    = (const float*)d_in[11];
    const float* n1_b1    = (const float*)d_in[12];
    const float* n1_w2    = (const float*)d_in[13];
    const float* n1_b2    = (const float*)d_in[14];
    const float* n1_ln_g  = (const float*)d_in[15];
    const float* n1_ln_b  = (const float*)d_in[16];
    const float* n2_res_w = (const float*)d_in[17];
    const float* n2_res_b = (const float*)d_in[18];
    const float* n2_w1    = (const float*)d_in[19];
    const float* n2_b1    = (const float*)d_in[20];
    const float* n2_w2    = (const float*)d_in[21];
    const float* n2_b2    = (const float*)d_in[22];
    const float* n2_ln_g  = (const float*)d_in[23];
    const float* n2_ln_b  = (const float*)d_in[24];
    const int* row = eidx;
    const int* col = eidx + NE;

    // --- workspace carve (all chunks 256B multiples) ---
    char* p = (char*)d_ws;
    auto take = [&](size_t bytes) -> char* { char* r = p; p += (bytes + 255) & ~size_t(255); return r; };
    _Float16* Wt_ew1  = (_Float16*)take((size_t)NL * HID * K_EW1 * 2);
    _Float16* Wt_n1w1 = (_Float16*)take((size_t)NL * HID * K_N1 * 2);
    _Float16* Wt_n1w2 = (_Float16*)take((size_t)NL * NPAD_N1W2 * HID * 2);
    _Float16* Wt_n2w1 = (_Float16*)take((size_t)NL * HID * K_EW1 * 2);
    _Float16* Wt_n2w2 = (_Float16*)take((size_t)NL * C * HID * 2);
    _Float16* Wt_n2rs = (_Float16*)take((size_t)NL * C * K_EW1 * 2);
    float*    nodes   = (float*)take((size_t)NN * C * 4);
    float*    edges   = (float*)take((size_t)NE * ECH * 4);
    float*    cnt     = (float*)take((size_t)NN * 4);
    _Float16* Xe      = (_Float16*)take((size_t)NE * K_EW1 * 2);
    _Float16* Hbuf    = (_Float16*)take((size_t)NE * HID * 2);   // reused for mlp1 hidden
    _Float16* Xn      = (_Float16*)take((size_t)NE * K_N1 * 2);
    float*    Mbuf    = (float*)take((size_t)NE * (C + ECH) * 4);
    float*    agg     = (float*)take((size_t)NN * (C + ECH) * 4);
    _Float16* Xn2     = (_Float16*)take((size_t)NN * K_EW1 * 2);
    _Float16* Hn      = (_Float16*)take((size_t)NN * HID * 2);
    float*    Tbuf    = (float*)take((size_t)NN * C * 4);

    const dim3 B256(256);
    // --- one-time weight prep ---
    prep_weight<<<dim3((HID * K_EW1 + 255) / 256, NL), B256, 0, stream>>>(e_w1,    Wt_ew1,  130, 128, K_EW1, HID);
    prep_weight<<<dim3((HID * K_N1  + 255) / 256, NL), B256, 0, stream>>>(n1_w1,   Wt_n1w1,  66, 128, K_N1,  HID);
    prep_weight<<<dim3((NPAD_N1W2 * HID + 255) / 256, NL), B256, 0, stream>>>(n1_w2, Wt_n1w2, 128, 66, HID, NPAD_N1W2);
    prep_weight<<<dim3((HID * K_EW1 + 255) / 256, NL), B256, 0, stream>>>(n2_w1,   Wt_n2w1, 130, 128, K_EW1, HID);
    prep_weight<<<dim3((C * HID + 255) / 256, NL), B256, 0, stream>>>(n2_w2,       Wt_n2w2, 128,  64, HID,   C);
    prep_weight<<<dim3((C * K_EW1 + 255) / 256, NL), B256, 0, stream>>>(n2_res_w,  Wt_n2rs, 130,  64, K_EW1, C);

    // --- state init ---
    init_nodes<<<((long)NN * C + 255) / 256, B256, 0, stream>>>(x, nodes);
    copy_f32<<<((long)NE * ECH + 255) / 256, B256, 0, stream>>>(eattr, edges, (long)NE * ECH);
    zero_f32<<<(NN + 255) / 256, B256, 0, stream>>>(cnt, NN);
    count_deg<<<(NE + 255) / 256, B256, 0, stream>>>(col, cnt);
    clamp_cnt<<<(NN + 255) / 256, B256, 0, stream>>>(cnt);

    const int MG_E = (NE / 16) / 8;  // 1792
    const int MG_N = (NN / 16) / 8;  // 256

    for (int l = 0; l < NL; ++l) {
        // EdgeModel
        build_edge_input<<<((long)NE * 32 + 255) / 256, B256, 0, stream>>>(nodes, edges, row, col, Xe);
        wmma_gemm<4, true, true, false><<<dim3(2, MG_E), B256, 0, stream>>>(
            Xe, K_EW1, Wt_ew1 + (long)l * HID * K_EW1, K_EW1, Hbuf, HID, HID, e_b1 + l * HID, K_EW1);
        edge_out<<<(NE + 255) / 256, B256, 0, stream>>>(
            Xe, Hbuf, e_w2 + (long)l * HID * ECH, e_b2 + l * ECH,
            e_res_w + (long)l * (2 * C + ECH) * ECH, e_res_b + l * ECH,
            e_ln_g + l * ECH, e_ln_b + l * ECH, edges, Xn);
        // NodeModel mlp_1
        wmma_gemm<4, true, true, false><<<dim3(2, MG_E), B256, 0, stream>>>(
            Xn, K_N1, Wt_n1w1 + (long)l * HID * K_N1, K_N1, Hbuf, HID, HID, n1_b1 + l * HID, K_N1);
        wmma_gemm<5, false, false, false><<<dim3(1, MG_E), B256, 0, stream>>>(
            Hbuf, HID, Wt_n1w2 + (long)l * NPAD_N1W2 * HID, HID, Mbuf, C + ECH, C + ECH,
            n1_b2 + l * (C + ECH), HID);
        zero_f32<<<((long)NN * (C + ECH) + 255) / 256, B256, 0, stream>>>(agg, (long)NN * (C + ECH));
        scatter_mean_ln<<<(NE + 255) / 256, B256, 0, stream>>>(
            Mbuf, Xn, col, n1_ln_g + l * (C + ECH), n1_ln_b + l * (C + ECH), agg);
        // NodeModel mlp_2
        build_node2_input<<<((long)NN * 32 + 255) / 256, B256, 0, stream>>>(nodes, agg, cnt, Xn2);
        wmma_gemm<4, true, true, false><<<dim3(2, MG_N), B256, 0, stream>>>(
            Xn2, K_EW1, Wt_n2w1 + (long)l * HID * K_EW1, K_EW1, Hn, HID, HID, n2_b1 + l * HID, K_EW1);
        wmma_gemm<4, false, false, false><<<dim3(1, MG_N), B256, 0, stream>>>(
            Hn, HID, Wt_n2w2 + (long)l * C * HID, HID, Tbuf, C, C, n2_b2 + l * C, HID);
        wmma_gemm<4, false, false, true><<<dim3(1, MG_N), B256, 0, stream>>>(
            Xn2, K_EW1, Wt_n2rs + (long)l * C * K_EW1, K_EW1, Tbuf, C, C, n2_res_b + l * C, K_EW1);
        node_out<<<(NN + 255) / 256, B256, 0, stream>>>(Tbuf, nodes, n2_ln_g + l * C, n2_ln_b + l * C);
    }

    copy_f32<<<((long)NA * C + 255) / 256, B256, 0, stream>>>(
        nodes + (long)(NN - NA) * C, (float*)d_out, (long)NA * C);
}
